// ScaledDotProductAttention_encoder_82978768158791
// MI455X (gfx1250) — matrix operand order repaired and checked by reading, not compiled, not run on hardware
//
#include <hip/hip_runtime.h>
#include <hip/hip_bf16.h>

// Problem constants (from reference): B=4, GRID=196, D_MODEL=512, H=8, D_K=D_V=64
#define B_   4
#define NQ   196
#define DM   512
#define H_   8
#define DK   64
#define MT   13      // ceil(196/16) row tiles

typedef __attribute__((ext_vector_type(16))) __bf16 v16bf;
typedef __attribute__((ext_vector_type(8)))  __bf16 v8bf;
typedef __attribute__((ext_vector_type(8)))  float  v8f;

__device__ __forceinline__ float sigmoidf_(float x) { return 1.0f / (1.0f + __expf(-x)); }

// ---------------------------------------------------------------------------
// fp32 -> bf16 conversion (grid-stride)
// ---------------------------------------------------------------------------
__global__ void f32_to_bf16_kernel(const float* __restrict__ src,
                                   __bf16* __restrict__ dst, int n) {
  int i = blockIdx.x * blockDim.x + threadIdx.x;
  int stride = gridDim.x * blockDim.x;
  for (; i < n; i += stride) dst[i] = (__bf16)src[i];
}

// ---------------------------------------------------------------------------
// One 16x32 (two adjacent 16x16 N-tiles) fp32 block of X(MxK) @ W^T,
// W row-major (N x K):  D[m][n] = sum_k X[m][k] * W[n][k]
// A fragment is loaded ONCE per K-step and fed to two WMMAs (A reuse).
// Wave32, v_wmma_f32_16x16x32_bf16, K stepped by 32.
// A layout (ISA 7.12.2, 16-bit A 16x32): lanes 0-15 row m=l16, K={0..7,16..23};
//                                        lanes 16-31 row m=l16, K={8..15,24..31}
// B layout (32x16): lane n<16 holds col n K={0..7,16..23}; lane n+16 col n K={8..15,24..31}
//   (col n of B == row n of W -> two contiguous 8-bf16 runs per lane)
// Rows/cols clamped for partial tiles (stores masked by the caller).
// ---------------------------------------------------------------------------
__device__ __forceinline__ void wmma_xwt_16x32(
    const __bf16* __restrict__ X, int ldx, int M, int mbase,
    const __bf16* __restrict__ Wt, int ldw, int N, int nbase, int Kdim,
    v8f& acc0, v8f& acc1)
{
  const int lane = threadIdx.x & 31;
  const int half = lane >> 4;
  const int l16  = lane & 15;
  int arow  = mbase + l16;      if (arow  > M - 1) arow  = M - 1;
  int bcol0 = nbase + l16;      if (bcol0 > N - 1) bcol0 = N - 1;
  int bcol1 = nbase + 16 + l16; if (bcol1 > N - 1) bcol1 = N - 1;
  const __bf16* ap  = X  + (size_t)arow  * ldx;
  const __bf16* bp0 = Wt + (size_t)bcol0 * ldw;
  const __bf16* bp1 = Wt + (size_t)bcol1 * ldw;
  acc0 = (v8f){};
  acc1 = (v8f){};
  for (int k0 = 0; k0 < Kdim; k0 += 32) {
    const int ka = k0 + half * 8;
    v8bf alo = *(const v8bf*)(ap + ka);        // K = ka .. ka+7
    v8bf ahi = *(const v8bf*)(ap + ka + 16);   // K = ka+16 .. ka+23
    const int kb = k0 + half * 8;
    v8bf b0a = *(const v8bf*)(bp0 + kb);
    v8bf b0b = *(const v8bf*)(bp0 + kb + 16);
    v8bf b1a = *(const v8bf*)(bp1 + kb);
    v8bf b1b = *(const v8bf*)(bp1 + kb + 16);
    v16bf a, b0, b1;
#pragma unroll
    for (int j = 0; j < 8; ++j) {
      a[j]  = alo[j]; a[j + 8]  = ahi[j];
      b0[j] = b0a[j]; b0[j + 8] = b0b[j];
      b1[j] = b1a[j]; b1[j + 8] = b1b[j];
    }
    acc0 = __builtin_amdgcn_wmma_f32_16x16x32_bf16(
        false, a, false, b0, (short)0, acc0, false, false);
    acc1 = __builtin_amdgcn_wmma_f32_16x16x32_bf16(
        false, a, false, b1, (short)0, acc1, false, false);
  }
}

// ---------------------------------------------------------------------------
// Q/K/V projections:  out = X @ W^T + bias   (one 16x32 block per wave)
// which: 0=Q (write fp32 + bf16), 1=K (fp32 + bf16), 2=V (fp32 only)
// ---------------------------------------------------------------------------
#define NPAIR_DM 16   // 512 / 32

__global__ void proj_kernel(const __bf16* __restrict__ qx, const __bf16* __restrict__ kx,
                            const __bf16* __restrict__ vx,
                            const __bf16* __restrict__ wq, const __bf16* __restrict__ wk,
                            const __bf16* __restrict__ wv,
                            const float* __restrict__ bq, const float* __restrict__ bk,
                            const float* __restrict__ bv,
                            float* __restrict__ qs, float* __restrict__ kp,
                            float* __restrict__ vp,
                            __bf16* __restrict__ qsb, __bf16* __restrict__ kpb)
{
  const int TPM = B_ * MT * NPAIR_DM;                 // blocks per matrix
  int gw = (blockIdx.x * blockDim.x + threadIdx.x) >> 5;
  if (gw >= 3 * TPM) return;                          // wave-uniform
  int which = gw / TPM;
  int r = gw % TPM;
  int b = r / (MT * NPAIR_DM); r %= (MT * NPAIR_DM);
  int mt = r / NPAIR_DM, np = r % NPAIR_DM;

  const __bf16* X    = (which == 0 ? qx : which == 1 ? kx : vx) + (size_t)b * NQ * DM;
  const __bf16* W    = (which == 0 ? wq : which == 1 ? wk : wv);
  const float*  bias = (which == 0 ? bq : which == 1 ? bk : bv);
  float*  outf = (which == 0 ? qs : which == 1 ? kp : vp);
  __bf16* outb = (which == 0 ? qsb : which == 1 ? kpb : (__bf16*)nullptr);

  v8f acc0, acc1;
  wmma_xwt_16x32(X, DM, NQ, mt * 16, W, DM, DM, np * 32, DM, acc0, acc1);

  const int lane = threadIdx.x & 31;
  const int half = lane >> 4, l16 = lane & 15;
  const int n0 = np * 32 + l16;
  const int n1 = n0 + 16;
  const float bn0 = bias[n0];
  const float bn1 = bias[n1];
#pragma unroll
  for (int i = 0; i < 8; ++i) {
    int m = mt * 16 + i + 8 * half;
    if (m < NQ) {
      size_t rowo = ((size_t)(b * NQ) + m) * DM;
      float v0 = acc0[i] + bn0;
      float v1 = acc1[i] + bn1;
      outf[rowo + n0] = v0;
      outf[rowo + n1] = v1;
      if (outb) { outb[rowo + n0] = (__bf16)v0; outb[rowo + n1] = (__bf16)v1; }
    }
  }
}

// ---------------------------------------------------------------------------
// Attention scores per (b,h): S = Qh(196x64) @ Kh^T(64x196), raw (pre-scale)
// 7 N-pairs cover 14 tiles (tile 13 fully masked out on store)
// ---------------------------------------------------------------------------
#define NPAIR_NQ 7

__global__ void score_kernel(const __bf16* __restrict__ qsb,
                             const __bf16* __restrict__ kpb,
                             float* __restrict__ S)
{
  int gw = (blockIdx.x * blockDim.x + threadIdx.x) >> 5;
  if (gw >= B_ * H_ * MT * NPAIR_NQ) return;          // wave-uniform
  int b = gw / (H_ * MT * NPAIR_NQ);
  int r = gw % (H_ * MT * NPAIR_NQ);
  int h = r / (MT * NPAIR_NQ); r %= (MT * NPAIR_NQ);
  int mt = r / NPAIR_NQ, np = r % NPAIR_NQ;

  const __bf16* A  = qsb + (size_t)b * NQ * DM + h * DK;  // rows of Qh
  const __bf16* Bm = kpb + (size_t)b * NQ * DM + h * DK;  // col n of Kh^T = row n of k_p
  v8f acc0, acc1;
  wmma_xwt_16x32(A, DM, NQ, mt * 16, Bm, DM, NQ, np * 32, DK, acc0, acc1);

  const int lane = threadIdx.x & 31;
  const int half = lane >> 4, l16 = lane & 15;
  const int n0 = np * 32 + l16;
  const int n1 = n0 + 16;
  float* Srow = S + ((size_t)(b * H_ + h) * NQ) * NQ;
#pragma unroll
  for (int i = 0; i < 8; ++i) {
    int m = mt * 16 + i + 8 * half;
    if (m < NQ) {
      if (n0 < NQ) Srow[(size_t)m * NQ + n0] = acc0[i];
      if (n1 < NQ) Srow[(size_t)m * NQ + n1] = acc1[i];
    }
  }
}

// ---------------------------------------------------------------------------
// Row softmax (in place) with per-element scale (reference passes full array)
// one block (256 threads) per (b,h,q) row of 196
// ---------------------------------------------------------------------------
__global__ void softmax_kernel(float* __restrict__ S, const float* __restrict__ scale)
{
  __shared__ float red[256];
  int row = blockIdx.x;               // over B_*H_*NQ
  int bh = row / NQ, q = row % NQ;
  int h = bh % H_;
  float* srow = S + (size_t)row * NQ;
  int t = threadIdx.x;

  float x = -3.4e38f;
  if (t < NQ) x = srow[t] * scale[((size_t)h * NQ + q) * NQ + t];
  red[t] = x;
  __syncthreads();
  for (int s = 128; s > 0; s >>= 1) {
    if (t < s) red[t] = fmaxf(red[t], red[t + s]);
    __syncthreads();
  }
  float mx = red[0];
  __syncthreads();
  float e = (t < NQ) ? __expf(x - mx) : 0.0f;
  red[t] = e;
  __syncthreads();
  for (int s = 128; s > 0; s >>= 1) {
    if (t < s) red[t] += red[t + s];
    __syncthreads();
  }
  float inv = 1.0f / red[0];
  if (t < NQ) srow[t] = e * inv;
}

// ---------------------------------------------------------------------------
// Sigmoid-gated weighted sum (cannot be a matmul: sigmoid sits inside the
// k-reduction):
//   image[b,q,h*64+d] = sum_k p[b,h,q,k] * sigmoid(qs_d * kp[k,d]) * vp[k,d]
// Block = (b,h, 4 q-rows); k_p/v_p head-tiles staged into LDS with CDNA5
// async copies (GLOBAL_LOAD_ASYNC_TO_LDS_B128, tracked by ASYNCcnt), shared
// by all 4 q-rows. blockDim = (64, 4); output written as bf16 for final GEMM.
// ---------------------------------------------------------------------------
__global__ void gated_sum_kernel(const float* __restrict__ qs,
                                 const float* __restrict__ kp,
                                 const float* __restrict__ vp,
                                 const float* __restrict__ p,
                                 __bf16* __restrict__ imgb)
{
  __shared__ float kpt[32][64];
  __shared__ float vpt[32][64];
  int blk = blockIdx.x;                       // over B_*H_*49
  int b = blk / (H_ * 49);
  int r = blk % (H_ * 49);
  int h = r / 49, qc = r % 49;
  int tx = threadIdx.x, ty = threadIdx.y;
  int q = qc * 4 + ty;                        // 196 = 49*4, always valid
  int hd = h * DK + tx;

  float qsd = qs[((size_t)(b * NQ) + q) * DM + hd];
  const float* prow = p + (((size_t)(b * H_ + h) * NQ) + q) * NQ;

  float acc = 0.0f;
  int flat = ty * 64 + tx;                    // 0..255
  for (int kb = 0; kb < NQ; kb += 32) {
    __syncthreads();                          // LDS reusable: prior reads done
    // Stage 32x64 fp32 tiles of kp and vp via async B128 copies.
    // Tile = 512 16-byte slots; 256 threads x 2 slots each per array.
#pragma unroll
    for (int rr = 0; rr < 2; ++rr) {
      int slot = flat + rr * 256;             // 0..511
      int kl = slot >> 4;                     // k row within tile (16 slots/row)
      int dl = (slot & 15) << 2;              // float offset within row
      int kg = kb + kl; if (kg > NQ - 1) kg = NQ - 1;
      size_t goff = ((size_t)(b * NQ) + kg) * DM + h * DK + dl;
      unsigned int lds_k = (unsigned int)(uintptr_t)&kpt[kl][dl];
      unsigned int lds_v = (unsigned int)(uintptr_t)&vpt[kl][dl];
      unsigned long long ga_k = (unsigned long long)(uintptr_t)(kp + goff);
      unsigned long long ga_v = (unsigned long long)(uintptr_t)(vp + goff);
      asm volatile("global_load_async_to_lds_b128 %0, %1, off"
                   :: "v"(lds_k), "v"(ga_k) : "memory");
      asm volatile("global_load_async_to_lds_b128 %0, %1, off"
                   :: "v"(lds_v), "v"(ga_v) : "memory");
    }
    asm volatile("s_wait_asynccnt 0x0" ::: "memory");
    __syncthreads();

    int kmax = (NQ - kb < 32) ? (NQ - kb) : 32;
    for (int kk = 0; kk < kmax; ++kk) {
      float pv = prow[kb + kk];
      float g = sigmoidf_(qsd * kpt[kk][tx]);
      acc += pv * g * vpt[kk][tx];
    }
  }
  imgb[((size_t)(b * NQ) + q) * DM + hd] = (__bf16)acc;
}

// ---------------------------------------------------------------------------
// Output projection + final gating:
//   pre = image @ Wo^T + bo;  out = sigmoid(q_sig * pre) * pre
// ---------------------------------------------------------------------------
__global__ void outproj_kernel(const __bf16* __restrict__ imgb,
                               const __bf16* __restrict__ wo,
                               const float* __restrict__ bo,
                               const float* __restrict__ qs,
                               float* __restrict__ out)
{
  int gw = (blockIdx.x * blockDim.x + threadIdx.x) >> 5;
  if (gw >= B_ * MT * NPAIR_DM) return;       // wave-uniform
  int b = gw / (MT * NPAIR_DM);
  int r = gw % (MT * NPAIR_DM);
  int mt = r / NPAIR_DM, np = r % NPAIR_DM;

  v8f acc0, acc1;
  wmma_xwt_16x32(imgb + (size_t)b * NQ * DM, DM, NQ, mt * 16,
                 wo, DM, DM, np * 32, DM, acc0, acc1);

  const int lane = threadIdx.x & 31;
  const int half = lane >> 4, l16 = lane & 15;
  const int n0 = np * 32 + l16;
  const int n1 = n0 + 16;
  const float bn0 = bo[n0];
  const float bn1 = bo[n1];
#pragma unroll
  for (int i = 0; i < 8; ++i) {
    int m = mt * 16 + i + 8 * half;
    if (m < NQ) {
      size_t rowo = ((size_t)(b * NQ) + m) * DM;
      float pre0 = acc0[i] + bn0;
      float pre1 = acc1[i] + bn1;
      float q0 = qs[rowo + n0];
      float q1 = qs[rowo + n1];
      out[rowo + n0] = sigmoidf_(q0 * pre0) * pre0;
      out[rowo + n1] = sigmoidf_(q1 * pre1) * pre1;
    }
  }
}

// ---------------------------------------------------------------------------
extern "C" void kernel_launch(void* const* d_in, const int* in_sizes, int n_in,
                              void* d_out, int out_size, void* d_ws, size_t ws_size,
                              hipStream_t stream)
{
  const float* queries = (const float*)d_in[0];
  const float* keys    = (const float*)d_in[1];
  const float* values  = (const float*)d_in[2];
  const float* Wq      = (const float*)d_in[3];
  const float* bq      = (const float*)d_in[4];
  const float* Wk      = (const float*)d_in[5];
  const float* bk      = (const float*)d_in[6];
  const float* Wv      = (const float*)d_in[7];
  const float* bv      = (const float*)d_in[8];
  const float* Wo      = (const float*)d_in[9];
  const float* bo      = (const float*)d_in[10];
  const float* scale   = (const float*)d_in[11];
  float* out = (float*)d_out;

  const size_t NE = (size_t)B_ * NQ * DM;   // 401408 activation elements
  const size_t NW = (size_t)DM * DM;        // 262144 weight elements
  const size_t NP = (size_t)B_ * H_ * NQ * NQ;

  char* ws = (char*)d_ws;
  size_t off = 0;
  auto alloc = [&](size_t bytes) -> char* {
    char* p = ws + off;
    off += (bytes + 255) & ~(size_t)255;
    return p;
  };
  __bf16* qx  = (__bf16*)alloc(NE * 2);
  __bf16* kx  = (__bf16*)alloc(NE * 2);
  __bf16* vx  = (__bf16*)alloc(NE * 2);
  __bf16* wqb = (__bf16*)alloc(NW * 2);
  __bf16* wkb = (__bf16*)alloc(NW * 2);
  __bf16* wvb = (__bf16*)alloc(NW * 2);
  __bf16* wob = (__bf16*)alloc(NW * 2);
  float*  qsf = (float*)alloc(NE * 4);
  float*  kpf = (float*)alloc(NE * 4);
  float*  vpf = (float*)alloc(NE * 4);
  __bf16* qsb = (__bf16*)alloc(NE * 2);
  __bf16* kpb = (__bf16*)alloc(NE * 2);
  float*  P   = (float*)alloc(NP * 4);
  __bf16* img = (__bf16*)alloc(NE * 2);
  (void)ws_size; (void)in_sizes; (void)n_in; (void)out_size;

  // 1) fp32 -> bf16 copies of activations and weights
  f32_to_bf16_kernel<<<(int)((NE + 255) / 256), 256, 0, stream>>>(queries, qx, (int)NE);
  f32_to_bf16_kernel<<<(int)((NE + 255) / 256), 256, 0, stream>>>(keys,    kx, (int)NE);
  f32_to_bf16_kernel<<<(int)((NE + 255) / 256), 256, 0, stream>>>(values,  vx, (int)NE);
  f32_to_bf16_kernel<<<(int)((NW + 255) / 256), 256, 0, stream>>>(Wq, wqb, (int)NW);
  f32_to_bf16_kernel<<<(int)((NW + 255) / 256), 256, 0, stream>>>(Wk, wkb, (int)NW);
  f32_to_bf16_kernel<<<(int)((NW + 255) / 256), 256, 0, stream>>>(Wv, wvb, (int)NW);
  f32_to_bf16_kernel<<<(int)((NW + 255) / 256), 256, 0, stream>>>(Wo, wob, (int)NW);

  // 2) Q/K/V projections (WMMA, A reuse x2): 3*B*13*16 = 2496 waves
  {
    int waves = 3 * B_ * MT * NPAIR_DM;
    proj_kernel<<<(waves + 3) / 4, 128, 0, stream>>>(
        qx, kx, vx, wqb, wkb, wvb, bq, bk, bv, qsf, kpf, vpf, qsb, kpb);
  }

  // 3) attention scores (WMMA): B*H*13*7 = 2912 waves
  {
    int waves = B_ * H_ * MT * NPAIR_NQ;
    score_kernel<<<(waves + 3) / 4, 128, 0, stream>>>(qsb, kpb, P);
  }

  // 4) softmax rows
  softmax_kernel<<<B_ * H_ * NQ, 256, 0, stream>>>(P, scale);

  // 5) sigmoid-gated weighted sum -> image (bf16), async LDS staging
  gated_sum_kernel<<<B_ * H_ * 49, dim3(64, 4), 0, stream>>>(qsf, kpf, vpf, P, img);

  // 6) output projection + final sigmoid gating (WMMA)
  {
    int waves = B_ * MT * NPAIR_DM;
    outproj_kernel<<<(waves + 3) / 4, 128, 0, stream>>>(img, wob, bo, qsf, out);
  }
}
